// AWDLSTM_9972914061862
// MI455X (gfx1250) — compile-verified, matrix operand order checked by reading, not run on hardware
//
#include <hip/hip_runtime.h>
#include <hip/hip_bf16.h>

#define T_LEN 1024
#define BATCH 64
#define DIN   512
#define HID   1024
#define KTOT  1536   // DIN + HID
#define NG    4096   // 4*HID
#define NWG   64     // 64 hidden slices of 16 columns
#define NTHR  256
#define KC    128    // K per chunk
#define NCH   12     // chunks per step (KC*NCH == KTOT)

typedef __attribute__((ext_vector_type(16))) __bf16 bf16x16;
typedef __attribute__((ext_vector_type(8)))  float  f32x8;

union Frag {
    uint4   u[2];
    bf16x16 v;
};

static __device__ __forceinline__ unsigned short f2bf(float f) {
    unsigned u = __float_as_uint(f);
    u = (u + 0x7FFFu + ((u >> 16) & 1u)) >> 16;   // round-to-nearest-even
    return (unsigned short)u;
}

// gfx1250 async copy: global memory -> LDS, 16B per lane, tracked by ASYNCcnt.
static __device__ __forceinline__ void async_copy_b128(unsigned lds_addr, const void* gptr) {
    asm volatile("global_load_async_to_lds_b128 %0, %1, off"
                 :: "v"(lds_addr), "v"(gptr)
                 : "memory");
}
// ASYNCcnt completes in order: waiting to <=4 means "my previous chunk landed".
#define WAIT_ASYNCCNT_4() asm volatile("s_wait_asynccnt 0x4" ::: "memory")
#define WAIT_ASYNCCNT_0() asm volatile("s_wait_asynccnt 0x0" ::: "memory")

// ---------------- pre-pass kernels ----------------

__global__ void init_h_kernel(const float* __restrict__ h0,
                              unsigned short* __restrict__ hbuf,
                              unsigned* __restrict__ counter) {
    size_t i = (size_t)blockIdx.x * blockDim.x + threadIdx.x;
    if (i == 0) *counter = 0u;
    if (i < (size_t)BATCH * HID) {
        hbuf[i] = f2bf(h0[i]);
        hbuf[(size_t)BATCH * HID + i] = 0;
    }
}

__global__ void pack_w_kernel(const float* __restrict__ Wi, const float* __restrict__ bi,
                              const float* __restrict__ Wh, const float* __restrict__ bh,
                              unsigned short* __restrict__ Wc, float* __restrict__ bsum) {
    size_t idx = (size_t)blockIdx.x * blockDim.x + threadIdx.x;
    if (idx < (size_t)NG * KTOT) {
        int n = (int)(idx / KTOT);
        int k = (int)(idx % KTOT);
        float v = (k < DIN) ? Wi[(size_t)n * DIN + k]
                            : Wh[(size_t)n * HID + (k - DIN)];
        Wc[idx] = f2bf(v);
    }
    if (idx < NG) bsum[idx] = bi[idx] + bh[idx];
}

__global__ void conv_x_kernel(const float* __restrict__ x,
                              unsigned short* __restrict__ xbf, size_t n) {
    size_t i = (size_t)blockIdx.x * blockDim.x + threadIdx.x;
    if (i < n) xbf[i] = f2bf(x[i]);
}

// ---------------- persistent LSTM kernel ----------------
// Grid: 64 WGs, WG s owns h-columns j in [s*16, s*16+16), full batch M=64.
// 8 waves: wave = (gate, m-half); one B fragment feeds 2 WMMAs.
// A (x_t||h) staged via double-buffered async DMA, pipelined against WMMA.
// Token-reset applied to h at WRITE time so staging is a pure copy.

__global__ __launch_bounds__(NTHR)
void lstm_persistent(const unsigned short* __restrict__ xbf,   // [T][B][DIN] bf16
                     const unsigned short* __restrict__ Wc,    // [NG][KTOT]  bf16 (K contiguous)
                     const float* __restrict__ bsum,           // [NG]
                     unsigned short* __restrict__ hbuf,        // [2][B][HID] bf16 ping-pong
                     const float* __restrict__ c0,             // [B][HID]
                     const int* __restrict__ tokens,           // [T][B]
                     float* __restrict__ out,                  // [T][B][HID] + hT + cT
                     unsigned* __restrict__ counter) {
    __shared__ unsigned short A_lds[2][64][KC];   // 2 x 16 KB double-buffered A chunk
    __shared__ float preact[4][16][64];           // 16 KB: gate, j-local, batch
    __shared__ float c_lds[16][64];               //  4 KB
    __shared__ float keep_lds[64];                // keep for THIS step's c
    __shared__ float keep2_lds[64];               // keep for NEXT step (h write mask)

    const int tid  = threadIdx.x;
    const int s    = blockIdx.x;       // hidden slice
    const int lane = tid & 31;
    const int wv   = tid >> 5;

    // init c slice from c0
    for (int cid = tid; cid < 1024; cid += NTHR) {
        int m = cid & 63, jl = cid >> 6;
        c_lds[jl][m] = c0[(size_t)m * HID + s * 16 + jl];
    }

    const int gate    = wv >> 1;                 // N-tile == gate
    const int mhalf   = wv & 1;                  // which pair of M-tiles
    const int colbase = gate * HID + s * 16;     // preact column base
    const int bn      = lane & 15;
    const int khalf   = (lane >> 4) * 8;
    const int arow0   = mhalf * 32 + bn;         // A rows for the two M-tiles
    const int arow1   = mhalf * 32 + 16 + bn;

    // staging addressing (4 async b128 per thread per chunk)
    const int sm  = (tid * 8) >> 7;              // batch row for pass 0
    const int skl = (tid * 8) & 127;             // col-in-chunk for pass 0

    for (int t = 0; t < T_LEN; ++t) {
        const int par = t & 1;

        f32x8 acc0 = {};
        f32x8 acc1 = {};

        __syncthreads();   // previous step done with A_lds / keep arrays
        if (tid < 64) {
            float kp = 1.f, kn = 1.f;
            if (t > 1 && tokens[(size_t)(t - 1) * BATCH + tid] == 0) kp = 0.f;
            if (t >= 1 && tokens[(size_t)t * BATCH + tid] == 0) kn = 0.f;
            keep_lds[tid]  = kp;   // masks c in this step's cell update
            keep2_lds[tid] = kn;   // masks h written for next step
        }

        // ---- issue chunk 0 ----
        #pragma unroll
        for (int p = 0; p < 4; ++p) {
            int m  = sm + p * 16;                // 16 rows per pass
            int kg = skl;                        // chunk 0: kg = kl < DIN always
            const unsigned short* gp = xbf + ((size_t)t * BATCH + m) * DIN + kg;
            async_copy_b128((unsigned)(uintptr_t)(&A_lds[0][m][skl]), gp);
        }

        #pragma unroll
        for (int c = 0; c < NCH; ++c) {
            // issue chunk c+1 into the other buffer (overlaps with compute of c)
            if (c + 1 < NCH) {
                const int kc = (c + 1) * KC;
                #pragma unroll
                for (int p = 0; p < 4; ++p) {
                    int m  = sm + p * 16;
                    int kg = kc + skl;
                    const unsigned short* gp =
                        (kg < DIN) ? (xbf + ((size_t)t * BATCH + m) * DIN + kg)
                                   : (hbuf + ((size_t)par * BATCH + m) * HID + (kg - DIN));
                    async_copy_b128((unsigned)(uintptr_t)(&A_lds[(c + 1) & 1][m][skl]), gp);
                }
                WAIT_ASYNCCNT_4();    // chunk c's 4 copies have landed (in-order)
            } else {
                WAIT_ASYNCCNT_0();
            }
            __syncthreads();          // chunk c visible to all waves

            const int bufc = c & 1;
            #pragma unroll
            for (int ki = 0; ki < 4; ++ki) {
                const int kb = ki * 32;
                Frag fa0, fa1, fb;
                fa0.u[0] = *(const uint4*)(&A_lds[bufc][arow0][kb + khalf]);
                fa0.u[1] = *(const uint4*)(&A_lds[bufc][arow0][kb + khalf + 16]);
                fa1.u[0] = *(const uint4*)(&A_lds[bufc][arow1][kb + khalf]);
                fa1.u[1] = *(const uint4*)(&A_lds[bufc][arow1][kb + khalf + 16]);
                const unsigned short* bp =
                    Wc + (size_t)(colbase + bn) * KTOT + (size_t)(c * KC + kb + khalf);
                fb.u[0] = *(const uint4*)(bp);
                fb.u[1] = *(const uint4*)(bp + 16);
                acc0 = __builtin_amdgcn_wmma_f32_16x16x32_bf16(
                           false, fa0.v, false, fb.v, (short)0, acc0, false, false);
                acc1 = __builtin_amdgcn_wmma_f32_16x16x32_bf16(
                           false, fa1.v, false, fb.v, (short)0, acc1, false, false);
            }
            __syncthreads();          // done reading buf c&1 before chunk c+2 overwrites it
        }

        // scatter preact tiles (+bias) into LDS by gate
        {
            const float bv = bsum[colbase + bn];
            const int mb = (lane >> 4) * 8;
            const int m0 = (mhalf * 2 + 0) * 16 + mb;
            const int m1 = (mhalf * 2 + 1) * 16 + mb;
            #pragma unroll
            for (int r = 0; r < 8; ++r) {
                preact[gate][bn][m0 + r] = acc0[r] + bv;
                preact[gate][bn][m1 + r] = acc1[r] + bv;
            }
        }
        __syncthreads();

        // cell update: 64 batch x 16 cols per WG
        for (int cid = tid; cid < 1024; cid += NTHR) {
            int m = cid & 63, jl = cid >> 6;
            float iv = preact[0][jl][m]; iv = 1.f / (1.f + __expf(-iv));
            float fv = preact[1][jl][m]; fv = 1.f / (1.f + __expf(-fv));
            float ov = preact[2][jl][m]; ov = 1.f / (1.f + __expf(-ov));
            float gv = tanhf(preact[3][jl][m]);
            float cold = c_lds[jl][m] * keep_lds[m];
            float cn = fv * cold + iv * gv;
            float hn = ov * tanhf(cn);
            c_lds[jl][m] = cn;
            int j = s * 16 + jl;
            // streaming output: non-temporal so it doesn't evict L2-resident weights
            __builtin_nontemporal_store(hn, &out[((size_t)t * BATCH + m) * HID + j]);
            // ping-pong h gets next step's reset pre-applied (pure copy at read)
            hbuf[((size_t)(par ^ 1) * BATCH + m) * HID + j] = f2bf(hn * keep2_lds[m]);
            if (t == T_LEN - 1) {
                __builtin_nontemporal_store(
                    hn, &out[(size_t)T_LEN * BATCH * HID + (size_t)m * HID + j]);           // hT
                __builtin_nontemporal_store(
                    cn, &out[(size_t)T_LEN * BATCH * HID + (size_t)BATCH * HID
                             + (size_t)m * HID + j]);                                       // cT
            }
        }

        // warm next step's x slab (64 WGs x 4 lines x 256B == full x_{t+1})
        if (tid < 4 && t + 1 < T_LEN) {
            __builtin_prefetch(xbf + (size_t)(t + 1) * BATCH * DIN
                                   + ((size_t)s * 4 + tid) * 128, 0, 3);
        }

        // grid-wide barrier (release h writes, spin on monotonic counter)
        __syncthreads();
        __threadfence();
        if (tid == 0) {
            atomicAdd(counter, 1u);
            const unsigned target = (unsigned)(t + 1) * NWG;
            while (*(volatile unsigned*)counter < target) {
                __builtin_amdgcn_s_sleep(1);
            }
        }
        __syncthreads();
        __threadfence();
    }
}

// ---------------- host launcher ----------------

extern "C" void kernel_launch(void* const* d_in, const int* in_sizes, int n_in,
                              void* d_out, int out_size, void* d_ws, size_t ws_size,
                              hipStream_t stream) {
    const float* x   = (const float*)d_in[0];   // [T,B,DIN]
    const float* h0  = (const float*)d_in[1];   // [B,H]
    const float* c0  = (const float*)d_in[2];   // [B,H]
    const float* Wi  = (const float*)d_in[3];   // [4H,DIN]
    const float* bi  = (const float*)d_in[4];   // [4H]
    const float* Wh  = (const float*)d_in[5];   // [4H,H]
    const float* bh  = (const float*)d_in[6];   // [4H]
    const int*   tok = (const int*)d_in[7];     // [T,B]
    float* out = (float*)d_out;

    // workspace layout
    char* base = (char*)d_ws;
    unsigned*       counter = (unsigned*)base;                                 // 256 B
    unsigned short* xbf     = (unsigned short*)(base + 256);                   // T*B*DIN bf16
    unsigned short* Wc      = xbf + (size_t)T_LEN * BATCH * DIN;               // NG*KTOT bf16
    float*          bsum    = (float*)(Wc + (size_t)NG * KTOT);               // NG f32
    unsigned short* hbuf    = (unsigned short*)(bsum + NG);                    // 2*B*H bf16
    (void)ws_size; (void)in_sizes; (void)n_in; (void)out_size;

    {
        size_t n = (size_t)BATCH * HID;
        init_h_kernel<<<(unsigned)((n + NTHR - 1) / NTHR), NTHR, 0, stream>>>(h0, hbuf, counter);
    }
    {
        size_t n = (size_t)NG * KTOT;
        pack_w_kernel<<<(unsigned)((n + NTHR - 1) / NTHR), NTHR, 0, stream>>>(Wi, bi, Wh, bh, Wc, bsum);
    }
    {
        size_t n = (size_t)T_LEN * BATCH * DIN;
        conv_x_kernel<<<(unsigned)((n + NTHR - 1) / NTHR), NTHR, 0, stream>>>(x, xbf, n);
    }
    lstm_persistent<<<NWG, NTHR, 0, stream>>>(xbf, Wc, bsum, hbuf, c0, tok, out, counter);
}
